// AttentionRelative_44160853738017
// MI455X (gfx1250) — matrix operand order repaired and checked by reading, not compiled, not run on hardware
//
#include <hip/hip_runtime.h>
#include <hip/hip_bf16.h>

// ---------------------------------------------------------------------------
// Problem constants (from reference)
// ---------------------------------------------------------------------------
#define BATCH   32
#define NPOS    1024
#define DIMX    512
#define HEADS   8
#define DHEAD   64
#define INNER   512          // HEADS*DHEAD
#define QKVW    1536         // 3*INNER
#define MROWS   32768        // BATCH*NPOS

typedef __attribute__((ext_vector_type(16))) __bf16 v16bf;
typedef __attribute__((ext_vector_type(8)))  __bf16 v8bf;
typedef __attribute__((ext_vector_type(8)))  float  v8f;
typedef __attribute__((ext_vector_type(4)))  unsigned int b128v;
typedef __attribute__((ext_vector_type(4)))  unsigned int u32x4;
typedef __attribute__((ext_vector_type(4)))  int  i32x4;
typedef __attribute__((ext_vector_type(8)))  int  i32x8;

__device__ __forceinline__ v16bf cat8(v8bf lo, v8bf hi) {
    return __builtin_shufflevector(lo, hi, 0,1,2,3,4,5,6,7,8,9,10,11,12,13,14,15);
}

__device__ __forceinline__ v8f wmma_bf16(v16bf a, v16bf b, v8f c) {
    return __builtin_amdgcn_wmma_f32_16x16x32_bf16(false, a, false, b, (short)0, c, false, false);
}

// ---------------------------------------------------------------------------
// Tensor Data Mover: async 2D bf16 tile load (global -> LDS), TENSORcnt-tracked.
// D# per CDNA5 ISA 8.3-8.6: group0 = count/lds_addr/global_addr/type=2,
// group1 = data_size=1 (2-byte units), dims/strides in elements; groups 2/3
// zero (tile_dim2 = 0 => 2D).
// ---------------------------------------------------------------------------
__device__ __forceinline__ void tdm_load_2d_bf16(unsigned lds_off,
                                                 const __bf16* gaddr,
                                                 unsigned row_elems,
                                                 unsigned rows,
                                                 unsigned row_stride_elems) {
    unsigned long long ga = (unsigned long long)(const void*)gaddr;
    u32x4 g0 = 0;
    g0[0] = 1u;                                            // count=1, user mode
    g0[1] = lds_off;                                       // lds_addr (bytes)
    g0[2] = (unsigned)(ga & 0xffffffffu);                  // global_addr[31:0]
    g0[3] = (unsigned)((ga >> 32) & 0x01ffffffu) | (2u << 30); // [56:32] | type=2
    i32x8 g1 = 0;
    g1[0] = 1 << 16;                                       // data_size=1 (2 bytes)
    g1[1] = (int)(row_elems << 16);                        // tensor_dim0[15:0]
    g1[2] = (int)(rows << 16);                             // tensor_dim1[15:0]
    g1[3] = (int)(row_elems << 16);                        // tile_dim0
    g1[4] = (int)rows;                                     // tile_dim1 (tile_dim2=0)
    g1[5] = (int)row_stride_elems;                         // tensor_dim0_stride[31:0]
    i32x4 z4 = 0;
#if __clang_major__ >= 23
    i32x8 z8 = 0;
    __builtin_amdgcn_tensor_load_to_lds(g0, g1, z4, z4, z8, 0);
#else
    __builtin_amdgcn_tensor_load_to_lds(g0, g1, z4, z4, 0);
#endif
}

// ---------------------------------------------------------------------------
// fp32 -> bf16 convert
// ---------------------------------------------------------------------------
__global__ void cvt_bf16_kernel(const float* __restrict__ src, __bf16* __restrict__ dst, int n) {
    int i = blockIdx.x * blockDim.x + threadIdx.x;
    if (i < n) dst[i] = (__bf16)src[i];
}

// ---------------------------------------------------------------------------
// Expand relative-position bias: biasf[h][i][j] = table[rel_index[i,j]*HEADS + h]
// ---------------------------------------------------------------------------
__global__ void bias_expand_kernel(const float* __restrict__ table,
                                   const int* __restrict__ ridx,
                                   float* __restrict__ biasf) {
    int ij = blockIdx.x * blockDim.x + threadIdx.x;
    if (ij >= NPOS * NPOS) return;
    int r = ridx[ij];
#pragma unroll
    for (int h = 0; h < HEADS; ++h)
        biasf[(size_t)h * NPOS * NPOS + ij] = table[r * HEADS + h];
}

// ---------------------------------------------------------------------------
// Tiled bf16 WMMA GEMM: C[M,N] = A[M,K] @ B[K,N]  (A,B bf16 row-major)
// 256 threads = 8 wave32 waves; block tile 128x128, k-step 32.
// Wave grid 2(m) x 4(n); each wave: 4 m-frags x 2 n-frags of 16x16.
// OUT_F32: C fp32 with +bias[col];  else C bf16.
// ---------------------------------------------------------------------------
template <bool OUT_F32>
__global__ void gemm_bf16_kernel(const __bf16* __restrict__ A,
                                 const __bf16* __restrict__ B,
                                 void* __restrict__ C,
                                 const float* __restrict__ bias,
                                 int M, int N, int K) {
    __shared__ __align__(16) __bf16 ldsA[128 * 32];   // [row][k]
    __shared__ __align__(16) __bf16 ldsBt[128 * 32];  // [col][k] (transposed)

    const int t    = threadIdx.x;
    const int wave = t >> 5;
    const int lane = t & 31;
    const int l16  = lane & 15;
    const int kg   = lane >> 4;        // 0/1 half of wave
    const int wm   = wave >> 2;        // 0..1
    const int wn   = wave & 3;         // 0..3
    const int mBlk = blockIdx.x * 128;
    const int nBlk = blockIdx.y * 128;

    v8f acc[4][2];
#pragma unroll
    for (int ms = 0; ms < 4; ++ms)
#pragma unroll
        for (int ns = 0; ns < 2; ++ns) acc[ms][ns] = 0.f;

    const int arow = t >> 1, aseg = (t & 1) * 16;   // A: 128 rows x 32 cols
    const int brow = t >> 3, bseg = (t & 7) * 16;   // B: 32 rows x 128 cols

    for (int k0 = 0; k0 < K; k0 += 32) {
        // ---- stage A tile (row-major) ----
        const __bf16* gA = A + (size_t)(mBlk + arow) * K + k0 + aseg;
        *(b128v*)&ldsA[arow * 32 + aseg]     = *(const b128v*)gA;
        *(b128v*)&ldsA[arow * 32 + aseg + 8] = *(const b128v*)(gA + 8);
        // ---- stage B tile transposed ----
        const __bf16* gB = B + (size_t)(k0 + brow) * N + nBlk + bseg;
        v8bf b0 = *(const v8bf*)gB;
        v8bf b1 = *(const v8bf*)(gB + 8);
#pragma unroll
        for (int e = 0; e < 8; ++e) {
            ldsBt[(bseg + e) * 32 + brow]     = b0[e];
            ldsBt[(bseg + 8 + e) * 32 + brow] = b1[e];
        }
        if (k0 + 32 < K) {
            __builtin_prefetch(gA + 32, 0, 3);
            __builtin_prefetch(gB + (size_t)32 * N, 0, 3);
        }
        __syncthreads();

#pragma unroll
        for (int ms = 0; ms < 4; ++ms) {
            // A fragment: lane row m=l16, k = kg*8+e (e<8) / 16+kg*8+(e-8)
            const __bf16* ap = &ldsA[(wm * 64 + ms * 16 + l16) * 32 + kg * 8];
            v16bf afrag = cat8(*(const v8bf*)ap, *(const v8bf*)(ap + 16));
#pragma unroll
            for (int ns = 0; ns < 2; ++ns) {
                // B fragment: lane col n=l16, k = kg*8+e (e<8) / 16+kg*8+(e-8)
                const __bf16* bp = &ldsBt[(wn * 32 + ns * 16 + l16) * 32 + kg * 8];
                v16bf bfrag = cat8(*(const v8bf*)bp, *(const v8bf*)(bp + 16));
                acc[ms][ns] = wmma_bf16(afrag, bfrag, acc[ms][ns]);
            }
        }
        __syncthreads();
    }

    // ---- epilogue: C row = base + v + 8*half, col = base + l16 ----
#pragma unroll
    for (int ms = 0; ms < 4; ++ms) {
#pragma unroll
        for (int ns = 0; ns < 2; ++ns) {
            const int col = nBlk + wn * 32 + ns * 16 + l16;
#pragma unroll
            for (int v = 0; v < 8; ++v) {
                const int row = mBlk + wm * 64 + ms * 16 + v + 8 * kg;
                const float val = acc[ms][ns][v];
                if (OUT_F32)
                    ((float*)C)[(size_t)row * N + col] = val + bias[col];
                else
                    ((__bf16*)C)[(size_t)row * N + col] = (__bf16)val;
            }
        }
    }
}

// ---------------------------------------------------------------------------
// Flash attention with relative bias.
// grid = (NPOS/128, BATCH*HEADS), block = 256 (8 waves).
// Each wave owns 16 q-rows; kv tiles of 64. All matmuls via WMMA bf16.
// Q tile lives in LDS (not registers) to avoid scratch spills; K tile staged
// by the Tensor Data Mover (async DMA, TENSORcnt); V tile staged transposed
// by the waves in parallel with the DMA.
// qkv layout: [MROWS, 1536] with cols [0,512)=Q, [512,1024)=K, [1024,1536)=V.
// ---------------------------------------------------------------------------
__global__ void flash_attn_kernel(const __bf16* __restrict__ qkv,
                                  const float* __restrict__ biasf,
                                  __bf16* __restrict__ attn_out) {
    __shared__ __align__(16) __bf16 ldsQ[128 * 64];       // [q_row][d]
    __shared__ __align__(16) __bf16 ldsK[64 * 64];        // [kv_row][d]  (TDM dest)
    __shared__ __align__(16) __bf16 ldsVt[64 * 64];       // [d][kv_row]
    __shared__ __align__(16) __bf16 ldsP[8 * 16 * 64];    // per-wave P strip

    const int t    = threadIdx.x;
    const int wave = t >> 5;
    const int lane = t & 31;
    const int l16  = lane & 15;
    const int kg   = lane >> 4;
    const int qt   = blockIdx.x;          // q tile (128 rows)
    const int bh   = blockIdx.y;
    const int b    = bh >> 3;
    const int h    = bh & 7;
    const float scale = 0.125f;           // DHEAD^-0.5
    const unsigned ldsK_off = (unsigned)(unsigned long long)(void*)&ldsK[0];

    // ---- stage whole Q tile (128 x 64) into LDS once ----
    {
        const int row = t >> 1;            // 0..127
        const int seg = (t & 1) * 32;      // 0 or 32
        const __bf16* qp = qkv + (size_t)(b * NPOS + qt * 128 + row) * QKVW + h * DHEAD + seg;
        *(b128v*)&ldsQ[row * 64 + seg]      = *(const b128v*)qp;
        *(b128v*)&ldsQ[row * 64 + seg + 8]  = *(const b128v*)(qp + 8);
        *(b128v*)&ldsQ[row * 64 + seg + 16] = *(const b128v*)(qp + 16);
        *(b128v*)&ldsQ[row * 64 + seg + 24] = *(const b128v*)(qp + 24);
    }

    v8f o[4];
#pragma unroll
    for (int ds = 0; ds < 4; ++ds) o[ds] = 0.f;
    float rmax[8], rsum[8];
#pragma unroll
    for (int v = 0; v < 8; ++v) { rmax[v] = -3.0e38f; rsum[v] = 0.f; }

    const float* bh_bias = biasf + (size_t)h * NPOS * NPOS;
    __bf16* pstrip = &ldsP[wave * 16 * 64];

    for (int kv0 = 0; kv0 < NPOS; kv0 += 64) {
        // ---- K tile via Tensor Data Mover (wave 0 issues; EXEC-independent) ----
        if (wave == 0) {
            tdm_load_2d_bf16(ldsK_off,
                             qkv + (size_t)(b * NPOS + kv0) * QKVW + 512 + h * DHEAD,
                             /*row_elems=*/DHEAD, /*rows=*/64,
                             /*row_stride_elems=*/QKVW);
        }
        // ---- stage V transposed (all waves, overlapped with the DMA) ----
        {
            const int j   = t >> 2;
            const int seg = (t & 3) * 16;
            const __bf16* vp = qkv + (size_t)(b * NPOS + kv0 + j) * QKVW + 1024 + h * DHEAD + seg;
            v8bf v0 = *(const v8bf*)vp;
            v8bf v1 = *(const v8bf*)(vp + 8);
#pragma unroll
            for (int e = 0; e < 8; ++e) {
                ldsVt[(seg + e) * 64 + j]     = v0[e];
                ldsVt[(seg + 8 + e) * 64 + j] = v1[e];
            }
        }
        if (wave == 0) __builtin_amdgcn_s_wait_tensorcnt(0);
        __syncthreads();

        // ---- Q A-fragments from LDS (short live range; no spills) ----
        v16bf qf[2];
#pragma unroll
        for (int kc = 0; kc < 2; ++kc) {
            const __bf16* qp = &ldsQ[(wave * 16 + l16) * 64 + kc * 32 + kg * 8];
            qf[kc] = cat8(*(const v8bf*)qp, *(const v8bf*)(qp + 16));
        }

        // ---- S = Q @ K^T  (4 n-subtiles, k-dim = 64 = 2 WMMA steps) ----
        v8f s[4];
#pragma unroll
        for (int ns = 0; ns < 4; ++ns) {
            v8f accs = 0.f;
#pragma unroll
            for (int kc = 0; kc < 2; ++kc) {
                // B fragment of K^T: lane col j = ns*16+l16, k(d) = kc*32 + kg*8 + e (e<8) / kc*32+16+kg*8+(e-8)
                const __bf16* kpp = &ldsK[(ns * 16 + l16) * 64 + kc * 32 + kg * 8];
                v16bf kfrag = cat8(*(const v8bf*)kpp, *(const v8bf*)(kpp + 16));
                accs = wmma_bf16(qf[kc], kfrag, accs);
            }
            s[ns] = accs;
        }

        // ---- scale + relative bias ----
#pragma unroll
        for (int ns = 0; ns < 4; ++ns) {
#pragma unroll
            for (int v = 0; v < 8; ++v) {
                const int gr = qt * 128 + wave * 16 + v + 8 * kg;
                const int gc = kv0 + ns * 16 + l16;
                s[ns][v] = s[ns][v] * scale + bh_bias[(size_t)gr * NPOS + gc];
            }
        }

        // ---- online softmax (rows are wave-private; reduce over 16 lanes) ----
        float mnew[8], corr[8];
#pragma unroll
        for (int v = 0; v < 8; ++v) {
            float m = fmaxf(fmaxf(s[0][v], s[1][v]), fmaxf(s[2][v], s[3][v]));
#pragma unroll
            for (int d = 1; d < 16; d <<= 1) m = fmaxf(m, __shfl_xor(m, d, 32));
            mnew[v] = fmaxf(rmax[v], m);
            corr[v] = __expf(rmax[v] - mnew[v]);
        }
#pragma unroll
        for (int ns = 0; ns < 4; ++ns)
#pragma unroll
            for (int v = 0; v < 8; ++v) s[ns][v] = __expf(s[ns][v] - mnew[v]);
#pragma unroll
        for (int v = 0; v < 8; ++v) {
            float ps = s[0][v] + s[1][v] + s[2][v] + s[3][v];
#pragma unroll
            for (int d = 1; d < 16; d <<= 1) ps += __shfl_xor(ps, d, 32);
            rsum[v] = rsum[v] * corr[v] + ps;
            rmax[v] = mnew[v];
        }
#pragma unroll
        for (int ds = 0; ds < 4; ++ds)
#pragma unroll
            for (int v = 0; v < 8; ++v) o[ds][v] *= corr[v];

        // ---- P (C-layout) -> LDS strip -> A-layout fragments ----
#pragma unroll
        for (int ns = 0; ns < 4; ++ns)
#pragma unroll
            for (int v = 0; v < 8; ++v)
                pstrip[(v + 8 * kg) * 64 + ns * 16 + l16] = (__bf16)s[ns][v];

        // ---- O += P @ V ----
#pragma unroll
        for (int kc = 0; kc < 2; ++kc) {
            const __bf16* pp = &pstrip[l16 * 64 + kc * 32 + kg * 8];     // A pattern
            v16bf pfrag = cat8(*(const v8bf*)pp, *(const v8bf*)(pp + 16));
#pragma unroll
            for (int ds = 0; ds < 4; ++ds) {
                const __bf16* vpp = &ldsVt[(ds * 16 + l16) * 64 + kc * 32 + kg * 8]; // B pattern
                v16bf vfrag = cat8(*(const v8bf*)vpp, *(const v8bf*)(vpp + 16));
                o[ds] = wmma_bf16(pfrag, vfrag, o[ds]);
            }
        }
        __syncthreads();   // protect K/V tiles before next staging
    }

    // ---- epilogue: normalize and store bf16 [b, n, h*64+d] ----
#pragma unroll
    for (int ds = 0; ds < 4; ++ds) {
#pragma unroll
        for (int v = 0; v < 8; ++v) {
            const int gr = qt * 128 + wave * 16 + v + 8 * kg;
            const int gc = h * DHEAD + ds * 16 + l16;
            attn_out[(size_t)(b * NPOS + gr) * INNER + gc] = (__bf16)(o[ds][v] / rsum[v]);
        }
    }
}

// ---------------------------------------------------------------------------
// Host-side launcher
// inputs: x, w_qkv, w_out, b_out, bias_table, rel_index
// ---------------------------------------------------------------------------
extern "C" void kernel_launch(void* const* d_in, const int* in_sizes, int n_in,
                              void* d_out, int out_size, void* d_ws, size_t ws_size,
                              hipStream_t stream) {
    const float* x     = (const float*)d_in[0];
    const float* w_qkv = (const float*)d_in[1];
    const float* w_out = (const float*)d_in[2];
    const float* b_out = (const float*)d_in[3];
    const float* btab  = (const float*)d_in[4];
    const int*   ridx  = (const int*)d_in[5];

    char* ws = (char*)d_ws;
    __bf16* xb    = (__bf16*)ws;  ws += (size_t)MROWS * DIMX  * 2;   // 32 MB
    __bf16* wqb   = (__bf16*)ws;  ws += (size_t)DIMX  * QKVW  * 2;   // 1.5 MB
    __bf16* wob   = (__bf16*)ws;  ws += (size_t)INNER * DIMX  * 2;   // 0.5 MB
    __bf16* qkv   = (__bf16*)ws;  ws += (size_t)MROWS * QKVW  * 2;   // 96 MB
    float*  biasf = (float*)ws;   ws += (size_t)HEADS * NPOS * NPOS * 4; // 32 MB
    __bf16* attnb = (__bf16*)ws;                                     // 32 MB

    // 1) precision downconvert (bf16 WMMA path)
    cvt_bf16_kernel<<<(MROWS * DIMX + 255) / 256, 256, 0, stream>>>(x, xb, MROWS * DIMX);
    cvt_bf16_kernel<<<(DIMX * QKVW + 255) / 256, 256, 0, stream>>>(w_qkv, wqb, DIMX * QKVW);
    cvt_bf16_kernel<<<(INNER * DIMX + 255) / 256, 256, 0, stream>>>(w_out, wob, INNER * DIMX);

    // 2) expand relative-position bias to [HEADS, N, N] fp32 (L2-resident)
    bias_expand_kernel<<<(NPOS * NPOS + 255) / 256, 256, 0, stream>>>(btab, ridx, biasf);

    // 3) QKV projection: [32768,512] @ [512,1536] -> bf16 qkv
    gemm_bf16_kernel<false><<<dim3(MROWS / 128, QKVW / 128), 256, 0, stream>>>(
        xb, wqb, qkv, nullptr, MROWS, QKVW, DIMX);

    // 4) fused flash attention with bias (WMMA + TDM)
    flash_attn_kernel<<<dim3(NPOS / 128, BATCH * HEADS), 256, 0, stream>>>(qkv, biasf, attnb);

    // 5) output projection: [32768,512] @ [512,512] + b_out -> fp32 d_out
    gemm_bf16_kernel<true><<<dim3(MROWS / 128, DIMX / 128), 256, 0, stream>>>(
        attnb, wob, d_out, b_out, MROWS, DIMX, DIMX);
}
